// VQ_VAE_25726854103591
// MI455X (gfx1250) — compile-verified, matrix operand order checked by reading, not run on hardware
//
#include <hip/hip_runtime.h>
#include <cstdint>

typedef __attribute__((ext_vector_type(16))) __bf16 v16bf;
typedef __attribute__((ext_vector_type(8)))  float  v8f;

union Bf16x8 { uint4 u; __bf16 b[8]; };
union FragU  { v16bf v; __bf16 e[16]; uint4 q[2]; };

#if __has_builtin(__builtin_amdgcn_global_load_async_to_lds_b128)
#define USE_ASYNC_LDS 1
// pointee type per clang prototype: int vector_size(16), AS1 (global) / AS3 (LDS)
typedef int v4i __attribute__((vector_size(16)));
typedef v4i __attribute__((address_space(1))) v4i_g;
typedef v4i __attribute__((address_space(3))) v4i_l;
#else
#define USE_ASYNC_LDS 0
#endif

static __device__ __forceinline__ __bf16 f2bf(float f) {
    // round-to-nearest-even f32 -> bf16
    unsigned u = __builtin_bit_cast(unsigned, f);
    u += 0x7fffu + ((u >> 16) & 1u);
    unsigned short h = (unsigned short)(u >> 16);
    return __builtin_bit_cast(__bf16, h);
}

template<bool GUARD>
static __device__ __forceinline__ void ld_chunk_f32(const float* rowp, int c, int K, __bf16* dst) {
    if (!GUARD || c < K) {
        float4 a = *(const float4*)(rowp + c);
        float4 b = *(const float4*)(rowp + c + 4);
        dst[0] = f2bf(a.x); dst[1] = f2bf(a.y); dst[2] = f2bf(a.z); dst[3] = f2bf(a.w);
        dst[4] = f2bf(b.x); dst[5] = f2bf(b.y); dst[6] = f2bf(b.z); dst[7] = f2bf(b.w);
    } else {
        #pragma unroll
        for (int j = 0; j < 8; ++j) dst[j] = f2bf(0.0f);
    }
}

template<bool GUARD>
static __device__ __forceinline__ void ld_chunk_bf16(const __bf16* rowp, int c, int K, __bf16* dst) {
    if (!GUARD || c < K) {
        *(uint4*)dst = *(const uint4*)(rowp + c);
    } else {
        uint4 z; z.x = z.y = z.z = z.w = 0u;
        *(uint4*)dst = z;
    }
}

// 16-byte global -> LDS staging; async on CDNA5 (ASYNCcnt), sync fallback.
static __device__ __forceinline__ void stage_b128(const __bf16* gsrc, __bf16* ldst) {
#if USE_ASYNC_LDS
    __builtin_amdgcn_global_load_async_to_lds_b128(
        (v4i_g*)(uintptr_t)gsrc,
        (v4i_l*)(uintptr_t)ldst,
        0, 0);
#else
    *(uint4*)ldst = *(const uint4*)gsrc;
#endif
}

template<int N>
static __device__ __forceinline__ void async_wait() {
#if USE_ASYNC_LDS
#if __has_builtin(__builtin_amdgcn_s_wait_asynccnt)
    __builtin_amdgcn_s_wait_asynccnt(N);
#else
    asm volatile("s_wait_asynccnt %0" :: "i"(N) : "memory");
#endif
#endif
}

// C(M,N) = act( A(M,K) @ Wb(Npad,Kpad)^T + bias ), M multiple of 128.
// Wb is pre-converted bf16, zero-padded to Npad x Kpad (Npad mult of 64, Kpad mult of 32).
// ACT: 0 = identity, 1 = relu, 2 = sigmoid.  ABF: A is bf16.  OBF: C is bf16.
template<int ACT, bool ABF, bool OBF>
__global__ __launch_bounds__(256)
void gemm_bias_act(const void* __restrict__ Av, const __bf16* __restrict__ Wb,
                   const float* __restrict__ bias, void* __restrict__ Cv,
                   int N, int K, int Kpad)
{
    constexpr int BSTR = 40; // bf16 row stride in LDS (80B rows => 16B aligned)
    __shared__ __align__(16) __bf16 Bs[2][64 * BSTR]; // double-buffered weight tiles

    const int tid  = threadIdx.x;
    const int lane = tid & 31;
    const int wave = tid >> 5;
    const int half = lane >> 4;
    const int l16  = lane & 15;

    const int mBase = blockIdx.y * 128 + wave * 16;
    const int nBase = blockIdx.x * 64;

    // B staging: thread t -> row (t>>2) of 64, k-chunk ((t&3)*8) of 32 (guard-free)
    const int srow = tid >> 2;
    const int scg  = (tid & 3) * 8;
    const __bf16* Wrow = Wb + (size_t)(nBase + srow) * (size_t)Kpad + scg;

    v8f acc[4];
    #pragma unroll
    for (int a = 0; a < 4; ++a) {
        v8f z = {0.f, 0.f, 0.f, 0.f, 0.f, 0.f, 0.f, 0.f};
        acc[a] = z;
    }

    const size_t aRowOff = (size_t)(mBase + l16) * (size_t)K;
    const float*  Af = (const float*)Av  + aRowOff;
    const __bf16* Ab = (const __bf16*)Av + aRowOff;

    auto stage_tile = [&](int s) {
        stage_b128(Wrow + s * 32, Bs[s & 1] + srow * BSTR + scg);
    };

    // A fragment for K-step s (16x32 bf16, ISA lane layout)
    auto load_a = [&](int s, bool guard) -> FragU {
        FragU fa;
        const int k0 = s * 32;
        if (guard) { // uniform branch: only the final K-step
            if constexpr (ABF) {
                ld_chunk_bf16<true>(Ab, k0 + half * 8,      K, fa.e);
                ld_chunk_bf16<true>(Ab, k0 + 16 + half * 8, K, fa.e + 8);
            } else {
                ld_chunk_f32<true> (Af, k0 + half * 8,      K, fa.e);
                ld_chunk_f32<true> (Af, k0 + 16 + half * 8, K, fa.e + 8);
            }
        } else {
            if constexpr (ABF) {
                ld_chunk_bf16<false>(Ab, k0 + half * 8,      K, fa.e);
                ld_chunk_bf16<false>(Ab, k0 + 16 + half * 8, K, fa.e + 8);
                __builtin_prefetch((const void*)(Ab + k0 + 64), 0, 1);
            } else {
                ld_chunk_f32<false> (Af, k0 + half * 8,      K, fa.e);
                ld_chunk_f32<false> (Af, k0 + 16 + half * 8, K, fa.e + 8);
                __builtin_prefetch((const void*)(Af + k0 + 64), 0, 1);
            }
        }
        return fa;
    };

    auto wmma4 = [&](const FragU& fa, const __bf16* Bbuf) {
        #pragma unroll
        for (int a = 0; a < 4; ++a) {
            FragU fb;
            const __bf16* Br = Bbuf + (a * 16 + l16) * BSTR;
            fb.q[0] = *(const uint4*)(Br + half * 8);
            fb.q[1] = *(const uint4*)(Br + 16 + half * 8);
            acc[a] = __builtin_amdgcn_wmma_f32_16x16x32_bf16(
                false, fa.v, false, fb.v, (short)0, acc[a], false, false);
        }
    };

    // ---- software-pipelined K loop: weight tile s+1 via async DMA and A
    // fragment s+1 via global loads are both in flight while computing step s.
    const int nsteps = (K + 31) >> 5;
    stage_tile(0);
    FragU fa_cur = load_a(0, nsteps == 1);
    int s = 0;
    for (; s + 1 < nsteps; ++s) {
        stage_tile(s + 1);                          // async DMA, in flight
        FragU fa_next = load_a(s + 1, s + 2 == nsteps);
        async_wait<1>();                            // tile s landed (in-order)
        __syncthreads();
        wmma4(fa_cur, Bs[s & 1]);
        __syncthreads();                            // all reads of Bs[s&1] done
        fa_cur = fa_next;
    }
    async_wait<0>();
    __syncthreads();
    wmma4(fa_cur, Bs[s & 1]);                       // final step

    // ---- epilogue: bias + activation + store ----
    #pragma unroll
    for (int a = 0; a < 4; ++a) {
        int n = nBase + a * 16 + l16;
        if (n < N) {
            float bv = bias[n];
            #pragma unroll
            for (int r = 0; r < 8; ++r) {
                int m = mBase + half * 8 + r;
                float v = acc[a][r] + bv;
                if (ACT == 1) v = fmaxf(v, 0.0f);
                if (ACT == 2) v = 1.0f / (1.0f + __expf(-v));
                if constexpr (OBF) ((__bf16*)Cv)[(size_t)m * N + n] = f2bf(v);
                else               ((float*)Cv) [(size_t)m * N + n] = v;
            }
        }
    }
}

// One-shot: convert W (N,K) f32 -> zero-padded bf16 panel (Npad,Kpad)
__global__ __launch_bounds__(256)
void convert_pad_w(const float* __restrict__ W, __bf16* __restrict__ wb,
                   int N, int K, int Kpad, int total /* Npad*Kpad */)
{
    int idx = blockIdx.x * 256 + threadIdx.x;
    if (idx >= total) return;
    int n = idx / Kpad;
    int k = idx - n * Kpad;
    float v = (k < K && n < N) ? W[(size_t)n * K + k] : 0.0f;
    wb[idx] = f2bf(v);
}

// Nearest-embed: z_e (B,10,20) flat (B,200), element [b,d,p] at b*200+d*20+p.
// emb (10,8) row-major. Writes z_q (B,200).
__global__ __launch_bounds__(256)
void vq_nearest(const float* __restrict__ ze, const float* __restrict__ emb,
                float* __restrict__ zq, int total /* B*20 */)
{
    __shared__ float se[80];
    if (threadIdx.x < 80) se[threadIdx.x] = emb[threadIdx.x];
    __syncthreads();

    int gid = blockIdx.x * 256 + threadIdx.x;
    if (gid >= total) return;
    int b = gid / 20;
    int p = gid - b * 20;

    const float* zp = ze + (size_t)b * 200 + p;
    float z[10];
    #pragma unroll
    for (int d = 0; d < 10; ++d) z[d] = zp[d * 20];

    int best = 0; float bd = 3.4e38f;
    #pragma unroll
    for (int k = 0; k < 8; ++k) {
        float dist = 0.f;
        #pragma unroll
        for (int d = 0; d < 10; ++d) { float df = z[d] - se[d * 8 + k]; dist += df * df; }
        if (dist < bd) { bd = dist; best = k; }
    }

    float* o = zq + (size_t)b * 200 + p;
    #pragma unroll
    for (int d = 0; d < 10; ++d) o[d * 20] = se[d * 8 + best];
}

extern "C" void kernel_launch(void* const* d_in, const int* in_sizes, int n_in,
                              void* d_out, int out_size, void* d_ws, size_t ws_size,
                              hipStream_t stream) {
    (void)n_in; (void)out_size; (void)ws_size;
    const float* x     = (const float*)d_in[0];
    const float* fc1_w = (const float*)d_in[1];
    const float* fc1_b = (const float*)d_in[2];
    const float* fc2_w = (const float*)d_in[3];
    const float* fc2_b = (const float*)d_in[4];
    const float* fc3_w = (const float*)d_in[5];
    const float* fc3_b = (const float*)d_in[6];
    const float* fc4_w = (const float*)d_in[7];
    const float* fc4_b = (const float*)d_in[8];
    const float* emb   = (const float*)d_in[9];

    const int B = in_sizes[0] / 784;   // 65536 (multiple of 128)

    float* out = (float*)d_out;                 // recon (B,784)
    float* ze  = out + (size_t)B * 784;         // z_e   (B,200)
    float* zq  = ze  + (size_t)B * 200;         // z_q   (B,200)

    // workspace layout (bf16 elements)
    __bf16* ws  = (__bf16*)d_ws;
    __bf16* h1b = ws;                    size_t off = (size_t)B * 400;   // (B,400)
    __bf16* h3b = ws + off;              off += (size_t)B * 400;         // (B,400)
    // padded bf16 weight panels: (Npad, Kpad)
    __bf16* wb1 = ws + off;              off += (size_t)448 * 800;       // 400x784 -> 448x800
    __bf16* wb2 = ws + off;              off += (size_t)256 * 416;       // 200x400 -> 256x416
    __bf16* wb3 = ws + off;              off += (size_t)448 * 224;       // 400x200 -> 448x224
    __bf16* wb4 = ws + off;              off += (size_t)832 * 416;       // 784x400 -> 832x416

    dim3 blk(256);

    // one-shot weight conversions (tiny)
    convert_pad_w<<<(448 * 800 + 255) / 256, blk, 0, stream>>>(fc1_w, wb1, 400, 784, 800, 448 * 800);
    convert_pad_w<<<(256 * 416 + 255) / 256, blk, 0, stream>>>(fc2_w, wb2, 200, 400, 416, 256 * 416);
    convert_pad_w<<<(448 * 224 + 255) / 256, blk, 0, stream>>>(fc3_w, wb3, 400, 200, 224, 448 * 224);
    convert_pad_w<<<(832 * 416 + 255) / 256, blk, 0, stream>>>(fc4_w, wb4, 784, 400, 416, 832 * 416);

    dim3 g1(448 / 64, B / 128);
    dim3 g2(256 / 64, B / 128);
    dim3 g3(448 / 64, B / 128);
    dim3 g4(832 / 64, B / 128);

    // h1 = relu(x @ fc1_w^T + b1)            f32 in -> bf16 out
    gemm_bias_act<1, false, true ><<<g1, blk, 0, stream>>>(x,   wb1, fc1_b, h1b, 400, 784, 800);
    // z_e = h1 @ fc2_w^T + b2                bf16 in -> f32 out (into d_out)
    gemm_bias_act<0, true,  false><<<g2, blk, 0, stream>>>(h1b, wb2, fc2_b, ze,  200, 400, 416);
    // z_q = nearest_embed(z_e)
    vq_nearest<<<(B * 20 + 255) / 256, 256, 0, stream>>>(ze, emb, zq, B * 20);
    // h3 = relu(z_q @ fc3_w^T + b3)          f32 in -> bf16 out
    gemm_bias_act<1, false, true ><<<g3, blk, 0, stream>>>(zq,  wb3, fc3_b, h3b, 400, 200, 224);
    // recon = sigmoid(h3 @ fc4_w^T + b4)     bf16 in -> f32 out (into d_out)
    gemm_bias_act<2, true,  false><<<g4, blk, 0, stream>>>(h3b, wb4, fc4_b, out, 784, 400, 416);
}
